// HIHI2_19653770346642
// MI455X (gfx1250) — matrix operand order repaired and checked by reading, not compiled, not run on hardware
//
#include <hip/hip_runtime.h>
#include <hip/hip_bf16.h>
#include <stdint.h>

// ---------------------------------------------------------------- types
typedef __attribute__((ext_vector_type(16))) __bf16 v16bf;
typedef __attribute__((ext_vector_type(8)))  float  v8f;
typedef int v4i_ __attribute__((vector_size(16)));   // matches async-LDS builtin param

#define PPOS   12544      // B*H*W = 16*28*28
#define HWIMG  784        // H*W
#define WIMG   28
#define DIMC   512
#define KCODE  2048
#define EPSV   1e-5f

// ---------------------------------------------------------------- CDNA5 async global->LDS
#if defined(__has_builtin)
#if __has_builtin(__builtin_amdgcn_global_load_async_to_lds_b128)
#define USE_ASYNC_LDS 1
#endif
#endif

#ifdef USE_ASYNC_LDS
__device__ __forceinline__ void async128(const unsigned short* g, unsigned short* l) {
  __builtin_amdgcn_global_load_async_to_lds_b128(
      (__attribute__((address_space(1))) v4i_*)(g),
      (__attribute__((address_space(3))) v4i_*)(l), 0, 0);
}
__device__ __forceinline__ void wait_async0() {
#if __has_builtin(__builtin_amdgcn_s_wait_asynccnt)
  __builtin_amdgcn_s_wait_asynccnt(0);
#else
  asm volatile("s_wait_asynccnt 0x0" ::: "memory");
#endif
}
#endif

// ---------------------------------------------------------------- helpers
__device__ __forceinline__ unsigned short f2bf(float f) {
  unsigned u = __float_as_uint(f);
  u += 0x7FFFu + ((u >> 16) & 1u);          // round-to-nearest-even
  return (unsigned short)(u >> 16);
}

__device__ __forceinline__ unsigned long long packScore(float s, unsigned n) {
  unsigned u = __float_as_uint(s);
  u = (u & 0x80000000u) ? ~u : (u | 0x80000000u);   // order-preserving float->uint
  return ((unsigned long long)u << 32) | (unsigned long long)(0xFFFFFFFFu - n);
}

__device__ __forceinline__ unsigned long long shflmax16(unsigned long long v) {
#pragma unroll
  for (int off = 1; off < 16; off <<= 1) {
    unsigned lo = (unsigned)v, hi = (unsigned)(v >> 32);
    unsigned olo = (unsigned)__shfl_xor((int)lo, off, 32);
    unsigned ohi = (unsigned)__shfl_xor((int)hi, off, 32);
    unsigned long long o = ((unsigned long long)ohi << 32) | olo;
    if (o > v) v = o;
  }
  return v;
}

// ---------------------------------------------------------------- tiny kernels
__global__ void k_zero_f32(float* p, int n) {
  int i = blockIdx.x * 256 + threadIdx.x; if (i < n) p[i] = 0.f;
}
__global__ void k_zero_u64(unsigned long long* p, int n) {
  int i = blockIdx.x * 256 + threadIdx.x; if (i < n) p[i] = 0ull;
}

__global__ void k_nchw2nhwc(const float* __restrict__ src, float* __restrict__ dst, int C, int n) {
  int i = blockIdx.x * 256 + threadIdx.x; if (i >= n) return;
  int p = i / C, c = i - p * C;
  int b = p / HWIMG, hw = p - b * HWIMG;
  dst[i] = src[((size_t)b * C + c) * HWIMG + hw];
}
__global__ void k_nhwc2nchw(const float* __restrict__ src, float* __restrict__ dst, int C, int n) {
  int i = blockIdx.x * 256 + threadIdx.x; if (i >= n) return;
  int b = i / (C * HWIMG); int rem = i - b * C * HWIMG;
  int c = rem / HWIMG; int hw = rem - c * HWIMG;
  dst[i] = src[((size_t)b * HWIMG + hw) * C + c];
}

__global__ void k_add2(const float* __restrict__ a, const float* __restrict__ b,
                       float* __restrict__ o, int n) {
  int i = blockIdx.x * 256 + threadIdx.x; if (i < n) o[i] = a[i] + b[i];
}

__global__ void k_to_bf16(const float* __restrict__ x, unsigned short* __restrict__ o, int n) {
  int i = blockIdx.x * 256 + threadIdx.x; if (i < n) o[i] = f2bf(x[i]);
}

// ---------------------------------------------------------------- BatchNorm (training stats)
__global__ void k_bn_partial(const float* __restrict__ x, float* __restrict__ sums, int C) {
  int p0 = blockIdx.x * 256;
  int c0 = threadIdx.x, c1 = threadIdx.x + 256;
  bool has1 = (c1 < C);
  float s0 = 0.f, q0 = 0.f, s1 = 0.f, q1 = 0.f;
  for (int pp = 0; pp < 256; ++pp) {
    const float* row = x + (size_t)(p0 + pp) * C;
    float v0 = row[c0]; s0 += v0; q0 += v0 * v0;
    if (has1) { float v1 = row[c1]; s1 += v1; q1 += v1 * v1; }
  }
  atomicAdd(&sums[c0], s0); atomicAdd(&sums[C + c0], q0);
  if (has1) { atomicAdd(&sums[c1], s1); atomicAdd(&sums[C + c1], q1); }
}

__global__ void k_bn_finalize(const float* __restrict__ sums, float* __restrict__ mean,
                              float* __restrict__ rstd, int C) {
  int c = blockIdx.x * 256 + threadIdx.x; if (c >= C) return;
  float m = sums[c] * (1.0f / PPOS);
  float v = sums[C + c] * (1.0f / PPOS) - m * m;
  mean[c] = m;
  rstd[c] = rsqrtf(v + EPSV);
}

__global__ void k_bn_apply_bf16(const float* __restrict__ x, const float* __restrict__ mean,
                                const float* __restrict__ rstd, const float* __restrict__ g,
                                const float* __restrict__ b, unsigned short* __restrict__ o,
                                int C, int relu, int n) {
  int i = blockIdx.x * 256 + threadIdx.x; if (i >= n) return;
  int c = i % C;
  float v = g[c] * (x[i] - mean[c]) * rstd[c] + b[c];
  if (relu) v = fmaxf(v, 0.f);
  o[i] = f2bf(v);
}

// ---------------------------------------------------------------- weight repack f32 -> bf16
// 3x3: OIHW -> n-major [co][(kh*3+kw)*Cin + ci]
__global__ void k_repack_w3(const float* __restrict__ w, unsigned short* __restrict__ wk,
                            int Cin, int Cout) {
  int i = blockIdx.x * 256 + threadIdx.x;
  if (i >= 9 * Cin * Cout) return;
  const int K3 = 9 * Cin;
  int co = i / K3, k = i - co * K3;
  int r = k / Cin, ci = k - r * Cin;
  wk[i] = f2bf(w[((size_t)co * Cin + ci) * 9 + r]);
}
// embed (DIM x K, k-contig) -> n-major [k][c]
__global__ void k_embed_t(const float* __restrict__ embed, unsigned short* __restrict__ et, int n) {
  int i = blockIdx.x * 256 + threadIdx.x; if (i >= n) return;
  int k = i / DIMC, c = i - k * DIMC;
  et[i] = f2bf(embed[(size_t)c * KCODE + k]);
}

// ---------------------------------------------------------------- WMMA fragment body
// LDS layout: AS[64][40] row-major (K contiguous), BT[64][40] n-major (K contiguous).
// A frag lane: K = {half*8 .. +7} and {16+half*8 .. +7}   -> 2x ds_load_b128
// B frag lane: K = {half*8 .. +7} and {16+half*8 .. +7} of column n -> 2x ds_load_b128
#define WMMA_BODY(AS, BT)                                                             \
  union { v16bf v; uint4 q[2]; } af;                                                  \
  {                                                                                   \
    const unsigned short* ap = &AS[wv * 16 + mrow][half * 8];                         \
    af.q[0] = *(const uint4*)ap;                                                      \
    af.q[1] = *(const uint4*)(ap + 16);                                               \
  }                                                                                   \
  _Pragma("unroll")                                                                   \
  for (int nf = 0; nf < 4; ++nf) {                                                    \
    union { v16bf v; uint4 q[2]; } bfr;                                               \
    const unsigned short* bp = &BT[nf * 16 + mrow][half * 8];                         \
    bfr.q[0] = *(const uint4*)bp;                                                     \
    bfr.q[1] = *(const uint4*)(bp + 16);                                              \
    acc[nf] = __builtin_amdgcn_wmma_f32_16x16x32_bf16(false, af.v, false, bfr.v,      \
                                                      (short)0, acc[nf], false, false); \
  }

// ---------------------------------------------------------------- generic GEMM (bf16 -> f32)
// C[M x N] = A[M x K] * Bn[N x K]^T (+ bias[n]).  M%64==0, N%64==0, K%32==0.
__global__ __launch_bounds__(128)
void k_gemm_bf16(const unsigned short* __restrict__ A, const unsigned short* __restrict__ Bn,
                 const float* __restrict__ bias, float* __restrict__ C, int N, int K) {
  __shared__ unsigned short As[2][64][40];
  __shared__ unsigned short Bt[2][64][40];
  const int tid = threadIdx.x;
  const int bn0 = blockIdx.x * 64, bm0 = blockIdx.y * 64;
  const int lane = tid & 31, wv = tid >> 5;
  const int mrow = lane & 15, half = lane >> 4;
  const int lr = tid >> 2, qc = (tid & 3) * 8;
  const unsigned short* aR0 = A + (size_t)(bm0 + lr) * K + qc;
  const unsigned short* aR1 = A + (size_t)(bm0 + lr + 32) * K + qc;
  const unsigned short* bR0 = Bn + (size_t)(bn0 + lr) * K + qc;
  const unsigned short* bR1 = Bn + (size_t)(bn0 + lr + 32) * K + qc;
  v8f acc[4] = {};
#ifdef USE_ASYNC_LDS
#define GISSUE(d, k0)                                 \
  do {                                                \
    async128(aR0 + (k0), &As[d][lr][qc]);             \
    async128(aR1 + (k0), &As[d][lr + 32][qc]);        \
    async128(bR0 + (k0), &Bt[d][lr][qc]);             \
    async128(bR1 + (k0), &Bt[d][lr + 32][qc]);        \
  } while (0)
#define GCOMMIT(d) do { wait_async0(); __syncthreads(); } while (0)
#else
  uint4 ra0, ra1, rb0, rb1;
#define GISSUE(d, k0)                                 \
  do {                                                \
    ra0 = *(const uint4*)(aR0 + (k0));                \
    ra1 = *(const uint4*)(aR1 + (k0));                \
    rb0 = *(const uint4*)(bR0 + (k0));                \
    rb1 = *(const uint4*)(bR1 + (k0));                \
  } while (0)
#define GCOMMIT(d)                                    \
  do {                                                \
    *(uint4*)&As[d][lr][qc] = ra0;                    \
    *(uint4*)&As[d][lr + 32][qc] = ra1;               \
    *(uint4*)&Bt[d][lr][qc] = rb0;                    \
    *(uint4*)&Bt[d][lr + 32][qc] = rb1;               \
    __syncthreads();                                  \
  } while (0)
#endif
  GISSUE(0, 0); GCOMMIT(0);
  int s = 0;
  for (int k0 = 32; k0 < K; k0 += 32) {
    GISSUE(s ^ 1, k0);                 // prefetch next tile
    { WMMA_BODY(As[s], Bt[s]) }        // compute current tile
    GCOMMIT(s ^ 1);                    // commit + one barrier
    s ^= 1;
  }
  { WMMA_BODY(As[s], Bt[s]) }
#undef GISSUE
#undef GCOMMIT
#pragma unroll
  for (int nf = 0; nf < 4; ++nf) {
    const int col = bn0 + nf * 16 + mrow;
    const float bv = bias ? bias[col] : 0.0f;
#pragma unroll
    for (int rr = 0; rr < 8; ++rr) {
      const int row = bm0 + wv * 16 + half * 8 + rr;
      C[(size_t)row * N + col] = acc[nf][rr] + bv;
    }
  }
}

// ---------------------------------------------------------------- 3x3 conv, implicit im2col
// A: NHWC bf16 (PPOS x Cin); Wk: n-major [Cout][9*Cin]; out: NHWC f32 (PPOS x Cout)
__global__ __launch_bounds__(128)
void k_conv3x3_bf16(const unsigned short* __restrict__ A, const unsigned short* __restrict__ Wk,
                    const float* __restrict__ bias, float* __restrict__ C,
                    int Cin, int Cout) {
  __shared__ unsigned short As[2][64][40];
  __shared__ unsigned short Bt[2][64][40];
  const int tid = threadIdx.x;
  const int bn0 = blockIdx.x * 64, bm0 = blockIdx.y * 64;
  const int lane = tid & 31, wv = tid >> 5;
  const int mrow = lane & 15, half = lane >> 4;
  const int Ktot = 9 * Cin;
  const int lr = tid >> 2, qc = (tid & 3) * 8;
  int bimg[2], py[2], px[2];
#pragma unroll
  for (int s2 = 0; s2 < 2; ++s2) {
    const int p = bm0 + lr + s2 * 32;
    bimg[s2] = p / HWIMG;
    const int hw = p - bimg[s2] * HWIMG;
    py[s2] = hw / WIMG;
    px[s2] = hw - py[s2] * WIMG;
  }
  const unsigned short* bR0 = Wk + (size_t)(bn0 + lr) * Ktot + qc;
  const unsigned short* bR1 = Wk + (size_t)(bn0 + lr + 32) * Ktot + qc;
  v8f acc[4] = {};
#ifdef USE_ASYNC_LDS
#define CISSUE(d, k0)                                                                  \
  do {                                                                                 \
    const int r_ = (k0) / Cin;                                                         \
    const int c0_ = (k0) - r_ * Cin;                                                   \
    const int dy_ = r_ / 3 - 1, dx_ = r_ - (r_ / 3) * 3 - 1;                           \
    _Pragma("unroll")                                                                  \
    for (int s2 = 0; s2 < 2; ++s2) {                                                   \
      const int y_ = py[s2] + dy_, x_ = px[s2] + dx_;                                  \
      if (y_ >= 0 && y_ < WIMG && x_ >= 0 && x_ < WIMG)                                \
        async128(A + (size_t)(bimg[s2] * HWIMG + y_ * WIMG + x_) * Cin + c0_ + qc,     \
                 &As[d][lr + s2 * 32][qc]);                                            \
      else                                                                             \
        *(uint4*)&As[d][lr + s2 * 32][qc] = make_uint4(0u, 0u, 0u, 0u);                \
    }                                                                                  \
    async128(bR0 + (k0), &Bt[d][lr][qc]);                                              \
    async128(bR1 + (k0), &Bt[d][lr + 32][qc]);                                         \
  } while (0)
#define CCOMMIT(d) do { wait_async0(); __syncthreads(); } while (0)
#else
  uint4 raA[2], rbB[2];
#define CISSUE(d, k0)                                                                  \
  do {                                                                                 \
    const int r_ = (k0) / Cin;                                                         \
    const int c0_ = (k0) - r_ * Cin;                                                   \
    const int dy_ = r_ / 3 - 1, dx_ = r_ - (r_ / 3) * 3 - 1;                           \
    _Pragma("unroll")                                                                  \
    for (int s2 = 0; s2 < 2; ++s2) {                                                   \
      const int y_ = py[s2] + dy_, x_ = px[s2] + dx_;                                  \
      raA[s2] = make_uint4(0u, 0u, 0u, 0u);                                            \
      if (y_ >= 0 && y_ < WIMG && x_ >= 0 && x_ < WIMG)                                \
        raA[s2] = *(const uint4*)(A + (size_t)(bimg[s2] * HWIMG + y_ * WIMG + x_) * Cin \
                                  + c0_ + qc);                                         \
    }                                                                                  \
    rbB[0] = *(const uint4*)(bR0 + (k0));                                              \
    rbB[1] = *(const uint4*)(bR1 + (k0));                                              \
  } while (0)
#define CCOMMIT(d)                                                                     \
  do {                                                                                 \
    *(uint4*)&As[d][lr][qc] = raA[0];                                                  \
    *(uint4*)&As[d][lr + 32][qc] = raA[1];                                             \
    *(uint4*)&Bt[d][lr][qc] = rbB[0];                                                  \
    *(uint4*)&Bt[d][lr + 32][qc] = rbB[1];                                             \
    __syncthreads();                                                                   \
  } while (0)
#endif
  CISSUE(0, 0); CCOMMIT(0);
  int s = 0;
  for (int k0 = 32; k0 < Ktot; k0 += 32) {
    CISSUE(s ^ 1, k0);
    { WMMA_BODY(As[s], Bt[s]) }
    CCOMMIT(s ^ 1);
    s ^= 1;
  }
  { WMMA_BODY(As[s], Bt[s]) }
#undef CISSUE
#undef CCOMMIT
#pragma unroll
  for (int nf = 0; nf < 4; ++nf) {
    const int col = bn0 + nf * 16 + mrow;
    const float bv = bias ? bias[col] : 0.0f;
#pragma unroll
    for (int rr = 0; rr < 8; ++rr) {
      const int row = bm0 + wv * 16 + half * 8 + rr;
      C[(size_t)row * Cout + col] = acc[nf][rr] + bv;
    }
  }
}

// ---------------------------------------------------------------- VQ score GEMM + fused argmax
// scores = F (PPOS x 512) * Et (2048 x 512)^T; argmax_n (score - 0.5*||e_n||^2)
__global__ __launch_bounds__(128)
void k_vq_score(const unsigned short* __restrict__ F, const unsigned short* __restrict__ Et,
                const float* __restrict__ enorm, unsigned long long* __restrict__ packed) {
  __shared__ unsigned short As[2][64][40];
  __shared__ unsigned short Bt[2][64][40];
  const int tid = threadIdx.x;
  const int bn0 = blockIdx.x * 64, bm0 = blockIdx.y * 64;
  const int lane = tid & 31, wv = tid >> 5;
  const int mrow = lane & 15, half = lane >> 4;
  const int lr = tid >> 2, qc = (tid & 3) * 8;
  const unsigned short* aR0 = F + (size_t)(bm0 + lr) * DIMC + qc;
  const unsigned short* aR1 = F + (size_t)(bm0 + lr + 32) * DIMC + qc;
  const unsigned short* bR0 = Et + (size_t)(bn0 + lr) * DIMC + qc;
  const unsigned short* bR1 = Et + (size_t)(bn0 + lr + 32) * DIMC + qc;
  v8f acc[4] = {};
#ifdef USE_ASYNC_LDS
#define VISSUE(d, k0)                                 \
  do {                                                \
    async128(aR0 + (k0), &As[d][lr][qc]);             \
    async128(aR1 + (k0), &As[d][lr + 32][qc]);        \
    async128(bR0 + (k0), &Bt[d][lr][qc]);             \
    async128(bR1 + (k0), &Bt[d][lr + 32][qc]);        \
  } while (0)
#define VCOMMIT(d) do { wait_async0(); __syncthreads(); } while (0)
#else
  uint4 ra0, ra1, rb0, rb1;
#define VISSUE(d, k0)                                 \
  do {                                                \
    ra0 = *(const uint4*)(aR0 + (k0));                \
    ra1 = *(const uint4*)(aR1 + (k0));                \
    rb0 = *(const uint4*)(bR0 + (k0));                \
    rb1 = *(const uint4*)(bR1 + (k0));                \
  } while (0)
#define VCOMMIT(d)                                    \
  do {                                                \
    *(uint4*)&As[d][lr][qc] = ra0;                    \
    *(uint4*)&As[d][lr + 32][qc] = ra1;               \
    *(uint4*)&Bt[d][lr][qc] = rb0;                    \
    *(uint4*)&Bt[d][lr + 32][qc] = rb1;               \
    __syncthreads();                                  \
  } while (0)
#endif
  VISSUE(0, 0); VCOMMIT(0);
  int s = 0;
  for (int k0 = 32; k0 < DIMC; k0 += 32) {
    VISSUE(s ^ 1, k0);
    { WMMA_BODY(As[s], Bt[s]) }
    VCOMMIT(s ^ 1);
    s ^= 1;
  }
  { WMMA_BODY(As[s], Bt[s]) }
#undef VISSUE
#undef VCOMMIT
#pragma unroll
  for (int rr = 0; rr < 8; ++rr) {
    unsigned long long best = 0ull;
#pragma unroll
    for (int nf = 0; nf < 4; ++nf) {
      const unsigned col = bn0 + nf * 16 + mrow;
      const float sc = acc[nf][rr] - 0.5f * enorm[col];
      const unsigned long long pk = packScore(sc, col);
      if (pk > best) best = pk;
    }
    best = shflmax16(best);           // reduce the 16 lanes sharing this row
    if (mrow == 0) {
      const int row = bm0 + wv * 16 + half * 8 + rr;
      atomicMax(&packed[row], best);
    }
  }
}

// ---------------------------------------------------------------- VQ helpers
__global__ void k_enorm(const float* __restrict__ embed, float* __restrict__ enorm) {
  const int k = blockIdx.x * 256 + threadIdx.x;
  if (k >= KCODE) return;
  float s = 0.f;
  for (int c = 0; c < DIMC; ++c) {
    const float v = embed[(size_t)c * KCODE + k];
    s += v * v;
  }
  enorm[k] = s;
}

__global__ void k_vq_gather(const unsigned long long* __restrict__ packed,
                            const float* __restrict__ embed, const float* __restrict__ fin,
                            float* __restrict__ resid, float* __restrict__ qsum,
                            float* __restrict__ diffacc, int accumulate) {
  __shared__ float red[256];
  const int i = blockIdx.x * 256 + threadIdx.x;       // over PPOS*512
  const int pidx = i >> 9;
  const int c = i & 511;
  const unsigned idxc = 0xFFFFFFFFu - (unsigned)(packed[pidx] & 0xFFFFFFFFull);
  const float q = embed[(size_t)c * KCODE + idxc];
  const float f = fin[i];
  resid[i] = f - q;
  if (accumulate) qsum[i] += q; else qsum[i] = q;
  const float d = q - f;
  red[threadIdx.x] = d * d;
  __syncthreads();
  for (int s = 128; s > 0; s >>= 1) {
    if (threadIdx.x < s) red[threadIdx.x] += red[threadIdx.x + s];
    __syncthreads();
  }
  if (threadIdx.x == 0) atomicAdd(diffacc, red[0]);
}

__global__ void k_write_diff(const float* __restrict__ acc, float* __restrict__ dst) {
  dst[0] = acc[0] * (1.0f / ((float)PPOS * (float)DIMC));   // Q_WEIGHT == 1.0
}

// ---------------------------------------------------------------- host orchestration
extern "C" void kernel_launch(void* const* d_in, const int* in_sizes, int n_in,
                              void* d_out, int out_size, void* d_ws, size_t ws_size,
                              hipStream_t stream) {
  (void)in_sizes; (void)n_in; (void)out_size; (void)ws_size;

  auto F = [&](int i) { return (const float*)d_in[i]; };
  const float* x = F(0);
  const float* enc0[11]; for (int j = 0; j < 11; ++j) enc0[j] = F(1 + j);
  const float* enc1[7];  for (int j = 0; j < 7;  ++j) enc1[j] = F(12 + j);
  const float* enc2[7];  for (int j = 0; j < 7;  ++j) enc2[j] = F(19 + j);
  const float* dec0[7];  for (int j = 0; j < 7;  ++j) dec0[j] = F(26 + j);
  const float* dec1[7];  for (int j = 0; j < 7;  ++j) dec1[j] = F(33 + j);
  const float* dec2[11]; for (int j = 0; j < 11; ++j) dec2[j] = F(40 + j);
  const float* vq1 = F(51); const float* vq2 = F(52); const float* vq3 = F(53);

  // workspace carve
  char* cur = (char*)d_ws;
  auto carve = [&](size_t bytes) { char* r = cur; cur += (bytes + 255) & ~(size_t)255; return r; };
  const size_t SB = sizeof(float) * (size_t)PPOS * DIMC;
  float* Z     = (float*)carve(SB);
  float* H1    = (float*)carve(SB);
  float* OUT2  = (float*)carve(SB);
  float* IDENT = (float*)carve(SB);
  float* QSUM  = (float*)carve(SB);
  float* RESID = (float*)carve(SB);
  unsigned short* ABF  = (unsigned short*)carve(2ull * PPOS * DIMC);
  unsigned short* WKBF = (unsigned short*)carve(2ull * 9 * DIMC * DIMC);
  unsigned short* EBF  = (unsigned short*)carve(2ull * DIMC * KCODE);
  float* SUMS  = (float*)carve(sizeof(float) * 2 * DIMC);
  float* MEAN  = (float*)carve(sizeof(float) * DIMC);
  float* RSTD  = (float*)carve(sizeof(float) * DIMC);
  float* ENORM = (float*)carve(sizeof(float) * KCODE);
  unsigned long long* PACKED = (unsigned long long*)carve(8ull * PPOS);
  float* DIFF  = (float*)carve(sizeof(float));

  auto bnstats = [&](const float* src, int C) {
    k_zero_f32<<<(2 * C + 255) / 256, 256, 0, stream>>>(SUMS, 2 * C);
    k_bn_partial<<<PPOS / 256, 256, 0, stream>>>(src, SUMS, C);
    k_bn_finalize<<<(C + 255) / 256, 256, 0, stream>>>(SUMS, MEAN, RSTD, C);
  };

  auto resblock = [&](const float* zin, int cin, int cout, const float* const* pp,
                      bool sc, float* zout) {
    const int nIn = PPOS * cin, nOut = PPOS * cout;
    bnstats(zin, cin);                       // BN1 / BNs share input batch stats
    if (sc) {
      k_bn_apply_bf16<<<nIn / 256, 256, 0, stream>>>(zin, MEAN, RSTD, pp[7], pp[8], ABF, cin, 0, nIn);
      // 1x1 OIHW is already n-major [co][ci]: plain f2bf copy
      k_to_bf16<<<(cin * cout + 255) / 256, 256, 0, stream>>>(pp[9], WKBF, cin * cout);
      k_gemm_bf16<<<dim3(cout / 64, PPOS / 64), 128, 0, stream>>>(ABF, WKBF, pp[10], IDENT, cout, cin);
    }
    k_bn_apply_bf16<<<nIn / 256, 256, 0, stream>>>(zin, MEAN, RSTD, pp[0], pp[1], ABF, cin, 1, nIn);
    k_repack_w3<<<(9 * cin * cin + 255) / 256, 256, 0, stream>>>(pp[2], WKBF, cin, cin);
    k_conv3x3_bf16<<<dim3(cin / 64, PPOS / 64), 128, 0, stream>>>(ABF, WKBF, nullptr, H1, cin, cin);
    bnstats(H1, cin);
    k_bn_apply_bf16<<<nIn / 256, 256, 0, stream>>>(H1, MEAN, RSTD, pp[3], pp[4], ABF, cin, 1, nIn);
    k_repack_w3<<<(9 * cin * cout + 255) / 256, 256, 0, stream>>>(pp[5], WKBF, cin, cout);
    k_conv3x3_bf16<<<dim3(cout / 64, PPOS / 64), 128, 0, stream>>>(ABF, WKBF, pp[6], OUT2, cin, cout);
    k_add2<<<nOut / 256, 256, 0, stream>>>(OUT2, sc ? IDENT : zin, zout, nOut);
  };

  auto vqprep = [&](const float* embed) {
    k_embed_t<<<(DIMC * KCODE) / 256, 256, 0, stream>>>(embed, EBF, DIMC * KCODE);
    k_enorm<<<KCODE / 256, 256, 0, stream>>>(embed, ENORM);
  };
  auto vqstage = [&](const float* embed, const float* fin, int accumulate) {
    k_to_bf16<<<(PPOS * DIMC) / 256, 256, 0, stream>>>(fin, ABF, PPOS * DIMC);
    k_zero_u64<<<(PPOS + 255) / 256, 256, 0, stream>>>(PACKED, PPOS);
    k_vq_score<<<dim3(KCODE / 64, PPOS / 64), 128, 0, stream>>>(ABF, EBF, ENORM, PACKED);
    k_vq_gather<<<(PPOS * DIMC) / 256, 256, 0, stream>>>(PACKED, embed, fin, RESID, QSUM, DIFF,
                                                         accumulate);
  };

  // ------- pipeline -------
  k_zero_f32<<<1, 256, 0, stream>>>(DIFF, 1);
  k_nchw2nhwc<<<(PPOS * 384) / 256, 256, 0, stream>>>(x, Z, 384, PPOS * 384);

  resblock(Z, 384, 512, enc0, true, RESID);     // encoder
  resblock(RESID, 512, 512, enc1, false, Z);
  resblock(Z, 512, 512, enc2, false, RESID);    // encoder output z -> RESID

  vqprep(vq1); vqstage(vq1, RESID, 0);          // q1 -> QSUM, r1 -> RESID
  vqprep(vq2); vqstage(vq2, RESID, 1);          // q1+q2, r2
  vqprep(vq3); vqstage(vq3, RESID, 1);          // q1+q2+q3

  resblock(QSUM, 512, 512, dec0, false, Z);     // decoder
  resblock(Z, 512, 512, dec1, false, RESID);
  resblock(RESID, 512, 384, dec2, true, Z);     // rec (NHWC) -> Z

  k_nhwc2nchw<<<(PPOS * 384) / 256, 256, 0, stream>>>(Z, (float*)d_out, 384, PPOS * 384);
  k_write_diff<<<1, 1, 0, stream>>>(DIFF, (float*)d_out + (size_t)PPOS * 384);
}
